// NSATransformerBlock_44409961840864
// MI455X (gfx1250) — matrix operand order repaired and checked by reading, not compiled, not run on hardware
//
#include <hip/hip_runtime.h>
#include <hip/hip_bf16.h>
#include <math.h>

typedef __attribute__((ext_vector_type(16))) _Float16 v16h;
typedef __attribute__((ext_vector_type(8)))  float    v8f;
typedef __attribute__((ext_vector_type(4)))  unsigned int v4u;
typedef __attribute__((ext_vector_type(8)))  int      v8i;
typedef __attribute__((ext_vector_type(4)))  int      v4i;

#define D_      1024
#define H_      16
#define HKV_    4
#define HD_     64
#define GQ_     4
#define B_      2
#define S_      2048
#define BS_     32
#define NB_     64
#define STRIDE_ 16
#define NC_     127
#define NCP_    128
#define TB_     16
#define LSEL_   512
#define WIN_    512
#define WK_     544
#define NEGV    (-1000000000.0f)
#define EPS_    1e-5f
#define SCALE_  0.125f
#define MROWS_  (B_*S_)

#define V8F_ZERO {0.f,0.f,0.f,0.f,0.f,0.f,0.f,0.f}

#if defined(__has_builtin)
#  if __has_builtin(__builtin_amdgcn_tensor_load_to_lds) && __has_builtin(__builtin_amdgcn_s_wait_tensorcnt)
#    define HAVE_TDM 1
#  else
#    define HAVE_TDM 0
#  endif
#else
#  define HAVE_TDM 0
#endif

__device__ __forceinline__ v8f wmma32(v16h a, v16h b, v8f c) {
    return __builtin_amdgcn_wmma_f32_16x16x32_f16(
        false, a, false, b, (short)0, c, false, false);
}

// A fragment, row-major f16 src (tile origin), leading dim ld.
// lane<16: M=lane, K=0..15; lane>=16: M=lane-16, K=16..31
__device__ __forceinline__ v16h frag_a_f16(const _Float16* __restrict__ src, int ld) {
    int lane = threadIdx.x & 31;
    int m = lane & 15, kh = (lane >> 4) * 16;
    const _Float16* p = src + (size_t)m * ld;
    v16h a;
#pragma unroll
    for (int i = 0; i < 8; ++i) { a[i] = p[kh + i]; a[8 + i] = p[8 + kh + i]; }
    return a;
}

// A fragment from f32 source (LDS probabilities), converted to f16.
__device__ __forceinline__ v16h frag_a_f32(const float* __restrict__ src, int ld) {
    int lane = threadIdx.x & 31;
    int m = lane & 15, kh = (lane >> 4) * 16;
    const float* p = src + (size_t)m * ld;
    v16h a;
#pragma unroll
    for (int i = 0; i < 8; ++i) { a[i] = (_Float16)p[kh + i]; a[8 + i] = (_Float16)p[8 + kh + i]; }
    return a;
}

// B fragment where B[k][n] = src[n][k] (transposed source; rows contiguous, ld = row stride)
// lane<16: N=lane, K=0..15; lane>=16: N=lane-16, K=16..31
__device__ __forceinline__ v16h frag_bt_f16(const _Float16* __restrict__ src, int ld) {
    int lane = threadIdx.x & 31;
    int n = lane & 15, k0 = (lane >> 4) * 16;
    const _Float16* p = src + (size_t)n * ld + k0;
    v16h b;
#pragma unroll
    for (int i = 0; i < 16; ++i) b[i] = p[i];
    return b;
}

// ------------------------------------------------------------------ TDM
#if HAVE_TDM
__device__ __forceinline__ void tdm_load_2d(const void* gaddr, unsigned lds_off,
                                            int tile_x, int tile_y,
                                            int stride_elems, int tensor_x, int tensor_y) {
    unsigned long long ga = (unsigned long long)(uintptr_t)gaddr;
    v4u g0;
    g0[0] = 1u;                                             // count=1, user mode
    g0[1] = lds_off;                                        // lds_addr (bytes)
    g0[2] = (unsigned)(ga & 0xFFFFFFFFu);                   // global_addr[31:0]
    g0[3] = (unsigned)((ga >> 32) & 0x1FFFFFFu) | (2u << 30); // addr[56:32], type=2
    v8i g1;
    g1[0] = (1 << 16);                                      // data_size=2B, mask=0
    g1[1] = (int)(((unsigned)tensor_x & 0xFFFFu) << 16);    // tensor_dim0[15:0]
    g1[2] = (int)((((unsigned)tensor_x >> 16) & 0xFFFFu) |
                  (((unsigned)tensor_y & 0xFFFFu) << 16));  // dim0[31:16] | dim1[15:0]
    g1[3] = (int)((((unsigned)tensor_y >> 16) & 0xFFFFu) |
                  (((unsigned)tile_x & 0xFFFFu) << 16));    // dim1[31:16] | tile_dim0
    g1[4] = (int)((unsigned)tile_y & 0xFFFFu);              // tile_dim1 (tile_dim2=0)
    g1[5] = stride_elems;                                   // tensor_dim0_stride[31:0]
    g1[6] = 0;
    g1[7] = 0;
    v4i z = {0, 0, 0, 0};
#if defined(__clang_major__) && (__clang_major__ >= 23)
    v8i z8 = {0, 0, 0, 0, 0, 0, 0, 0};
    __builtin_amdgcn_tensor_load_to_lds(g0, g1, z, z, z8, 0);
#else
    __builtin_amdgcn_tensor_load_to_lds(g0, g1, z, z, 0);
#endif
}
#endif

// ---------------------------------------------------------------- utilities
// transpose+convert: src f32 [K][N] -> dst f16 [N][K]
__global__ void cvt_tr(const float* __restrict__ src, _Float16* __restrict__ dst,
                       int K, int N) {
    __shared__ float t[32][33];
    int k0 = blockIdx.x * 32, n0 = blockIdx.y * 32;
    int tx = threadIdx.x & 31, ty = threadIdx.x >> 5;
    for (int r = ty; r < 32; r += 8)
        if (n0 + tx < N) t[r][tx] = src[(size_t)(k0 + r) * N + n0 + tx];
    __syncthreads();
    for (int r = ty; r < 32; r += 8)
        if (n0 + r < N) dst[(size_t)(n0 + r) * K + k0 + tx] = (_Float16)t[tx][r];
}

// V16 [B,S,HKV,HD] -> VT [B][HKV][HD][S]
__global__ void v_transpose(const _Float16* __restrict__ v16, _Float16* __restrict__ vt) {
    int idx = blockIdx.x * 256 + threadIdx.x;
    int s = idx & (S_ - 1);
    int d = (idx >> 11) & (HD_ - 1);
    int bk = idx >> 17;
    int b = bk >> 2, kv = bk & 3;
    vt[((size_t)bk * HD_ + d) * S_ + s] =
        v16[(size_t)(b * S_ + s) * (HKV_ * HD_) + kv * HD_ + d];
}

// LayerNorm over D=1024, f32 in -> f16 out. One block (256 thr) per row.
__global__ void ln_f16(const float* __restrict__ x, const float* __restrict__ w,
                       const float* __restrict__ b, _Float16* __restrict__ out) {
    __shared__ float red[256];
    int r = blockIdx.x;
    const float* xr = x + (size_t)r * D_;
    float s = 0.f;
    for (int i = threadIdx.x; i < D_; i += 256) s += xr[i];
    red[threadIdx.x] = s; __syncthreads();
    for (int st = 128; st > 0; st >>= 1) {
        if (threadIdx.x < st) red[threadIdx.x] += red[threadIdx.x + st];
        __syncthreads();
    }
    float mean = red[0] * (1.f / D_);
    __syncthreads();
    float v = 0.f;
    for (int i = threadIdx.x; i < D_; i += 256) { float d = xr[i] - mean; v += d * d; }
    red[threadIdx.x] = v; __syncthreads();
    for (int st = 128; st > 0; st >>= 1) {
        if (threadIdx.x < st) red[threadIdx.x] += red[threadIdx.x + st];
        __syncthreads();
    }
    float inv = rsqrtf(red[0] * (1.f / D_) + EPS_);
    for (int i = threadIdx.x; i < D_; i += 256)
        out[(size_t)r * D_ + i] = (_Float16)((xr[i] - mean) * inv * w[i] + b[i]);
}

// -------------------------------------------------------------- WMMA GEMM
// C[M][N] = A16[M][K] @ BT16[N][K]^T + bias.
// Block: 128 thr = 4 waves, tile M=64 x N=64 (wave w owns n-tile w, 4 m-tiles).
// A panel (64 x 256) staged to LDS by the Tensor Data Mover, shared by all waves.
// Epilogues: 0 f32 | 1 f16 | 2 f16+GELU | 3 f32+sigmoid | 4 f32+residual
template <int EP>
__global__ __launch_bounds__(128) void gemm16(
        const _Float16* __restrict__ A, const _Float16* __restrict__ BT,
        const float* __restrict__ bias, int M, int N, int K,
        void* __restrict__ outp, const float* __restrict__ res) {
    __shared__ _Float16 As[64 * 256];   // 32 KB
    int lane = threadIdx.x & 31;
    int wid  = threadIdx.x >> 5;
    int m0 = blockIdx.x * 64;
    int n0 = (blockIdx.y * 4 + wid) * 16;
    bool active = (n0 < N);
    int mrow = lane & 15, khalf = (lane >> 4) * 16;
    int ncol = lane & 15, mb = (lane >> 4) * 8;
    const _Float16* btrow = BT + (size_t)(n0 + ncol) * K;
    v8f acc[4] = {V8F_ZERO, V8F_ZERO, V8F_ZERO, V8F_ZERO};
    for (int kk = 0; kk < K; kk += 256) {
        __syncthreads();                      // LDS reuse WAR
#if HAVE_TDM
        if (wid == 0) {
            tdm_load_2d(A + (size_t)m0 * K + kk, (unsigned)(uintptr_t)As,
                        /*tile_x=*/256, /*tile_y=*/64, /*stride=*/K,
                        /*tensor_x=*/K, /*tensor_y=*/M);
            __builtin_amdgcn_s_wait_tensorcnt((short)0);
        }
#else
        for (int c = threadIdx.x; c < 2048; c += 128) {
            int row = c >> 5, col = (c & 31) * 8;
            *(uint4*)(As + row * 256 + col) =
                *(const uint4*)(A + (size_t)(m0 + row) * K + kk + col);
        }
#endif
        __syncthreads();
        if (active) {
            for (int k0 = 0; k0 < 256; k0 += 32) {
                const _Float16* bp = btrow + kk + k0 + (lane >> 4) * 16;
                __builtin_prefetch(bp + 256, 0, 1);
                v16h bf;
#pragma unroll
                for (int i = 0; i < 16; ++i) bf[i] = bp[i];
#pragma unroll
                for (int t = 0; t < 4; ++t) {
                    const _Float16* ap = As + (t * 16 + mrow) * 256 + k0;
                    v16h af;
#pragma unroll
                    for (int i = 0; i < 8; ++i) { af[i] = ap[khalf + i]; af[8 + i] = ap[8 + khalf + i]; }
                    acc[t] = wmma32(af, bf, acc[t]);
                }
            }
        }
    }
    if (!active) return;
#pragma unroll
    for (int t = 0; t < 4; ++t) {
#pragma unroll
        for (int r = 0; r < 8; ++r) {
            int m = m0 + t * 16 + mb + r, n = n0 + ncol;
            float vv = acc[t][r] + bias[n];
            size_t o = (size_t)m * N + n;
            if (EP == 0) {
                ((float*)outp)[o] = vv;
            } else if (EP == 1) {
                ((_Float16*)outp)[o] = (_Float16)vv;
            } else if (EP == 2) {
                float g = 0.5f * vv * (1.f + erff(vv * 0.70710678118654752f));
                ((_Float16*)outp)[o] = (_Float16)g;
            } else if (EP == 3) {
                ((float*)outp)[o] = 1.f / (1.f + expf(-vv));
            } else {
                ((float*)outp)[o] = vv + res[o];
            }
        }
    }
}

// ------------------------------------------------- compressed K/V pooling
// ck [B*HKV][NCP][HD] row-major ; cvt [B*HKV][HD][NCP] transposed ; pad rows zero
__global__ void compress_kv(const _Float16* __restrict__ k16, const _Float16* __restrict__ v16,
                            _Float16* __restrict__ ck, _Float16* __restrict__ cvt) {
    int j = blockIdx.x, bk = blockIdx.y, d = threadIdx.x;
    int b = bk >> 2, kv = bk & 3;
    size_t ok = ((size_t)bk * NCP_ + j) * HD_ + d;
    size_t ov = ((size_t)bk * HD_ + d) * NCP_ + j;
    if (j >= NC_) { ck[ok] = (_Float16)0.f; cvt[ov] = (_Float16)0.f; return; }
    float sk = 0.f, sv = 0.f;
    for (int i = 0; i < BS_; ++i) {
        size_t ro = (size_t)(b * S_ + j * STRIDE_ + i) * (HKV_ * HD_) + kv * HD_ + d;
        sk += (float)k16[ro];
        sv += (float)v16[ro];
    }
    ck[ok] = (_Float16)(sk * (1.f / BS_));
    cvt[ov] = (_Float16)(sv * (1.f / BS_));
}

__global__ void block_means(const _Float16* __restrict__ q16, const _Float16* __restrict__ k16,
                            float* __restrict__ qsum, float* __restrict__ kbm) {
    int n = blockIdx.x, bk = blockIdx.y, d = threadIdx.x;
    int b = bk >> 2, kv = bk & 3;
    float sk = 0.f, sq = 0.f;
    for (int i = 0; i < BS_; ++i) {
        int s = n * BS_ + i;
        sk += (float)k16[(size_t)(b * S_ + s) * (HKV_ * HD_) + kv * HD_ + d];
        for (int g = 0; g < GQ_; ++g)
            sq += (float)q16[(size_t)(b * S_ + s) * (H_ * HD_) + (kv * GQ_ + g) * HD_ + d];
    }
    size_t o = ((size_t)bk * NB_ + n) * HD_ + d;
    kbm[o]  = sk * (1.f / BS_);
    qsum[o] = sq * (1.f / BS_);
}

__global__ void imp_topk(const float* __restrict__ qsum, const float* __restrict__ kbm,
                         int* __restrict__ sel) {
    int bk = blockIdx.x;
    int n  = threadIdx.x;
    float qv[HD_];
    for (int d = 0; d < HD_; ++d) qv[d] = qsum[((size_t)bk * NB_ + n) * HD_ + d];
    float vals[NB_];
    for (int m = 0; m < NB_; ++m) {
        const float* kr = kbm + ((size_t)bk * NB_ + m) * HD_;
        float a = 0.f;
        for (int d = 0; d < HD_; ++d) a += qv[d] * kr[d];
        if (m > n) a = NEGV;
        if (m == n) a = 1000000000.0f;
        vals[m] = a;
    }
    unsigned long long used = 0ull;
    for (int t = 0; t < TB_; ++t) {
        float best = -INFINITY; int bi = 0;
        for (int m = 0; m < NB_; ++m)
            if (!((used >> m) & 1ull) && vals[m] > best) { best = vals[m]; bi = m; }
        used |= 1ull << bi;
        sel[((size_t)bk * NB_ + n) * TB_ + t] = bi;
    }
}

// -------------------------------------------------- compressed attention
__global__ void attn_cmp(const _Float16* __restrict__ q16, const _Float16* __restrict__ ck,
                         const _Float16* __restrict__ cvt, float* __restrict__ oc) {
    __shared__ float sc[16][NCP_];
    int lane = threadIdx.x;
    int m0 = blockIdx.x * 16;
    int bh = blockIdx.y, b = bh >> 4, h = bh & 15, kv = h >> 2;
    int mrow = lane & 15, khalf = (lane >> 4) * 8;
    int ncol = lane & 15, mb = (lane >> 4) * 8;
    const _Float16* qtile = q16 + (size_t)(b * S_ + m0) * (H_ * HD_) + h * HD_;
    const _Float16* ckb  = ck  + (size_t)(b * HKV_ + kv) * NCP_ * HD_;
    const _Float16* cvtb = cvt + (size_t)(b * HKV_ + kv) * HD_ * NCP_;
    for (int nt = 0; nt < NCP_ / 16; ++nt) {
        v8f acc = V8F_ZERO;
        for (int k0 = 0; k0 < HD_; k0 += 32) {
            v16h af = frag_a_f16(qtile + k0, H_ * HD_);
            v16h bf = frag_bt_f16(ckb + (size_t)(nt * 16) * HD_ + k0, HD_);
            acc = wmma32(af, bf, acc);
        }
#pragma unroll
        for (int r = 0; r < 8; ++r) sc[mb + r][nt * 16 + ncol] = acc[r];
    }
    __syncthreads();
    if (lane < 16) {
        int s = m0 + lane;
        float mx = -INFINITY; bool any = false;
        for (int j = 0; j < NCP_; ++j) {
            float v;
            if (j < NC_ && s >= j * STRIDE_ + BS_ - 1) { v = sc[lane][j] * SCALE_; any = true; }
            else v = NEGV;
            sc[lane][j] = v;
            if (v > mx) mx = v;
        }
        float sum = 0.f;
        for (int j = 0; j < NCP_; ++j) {
            float e = any ? expf(sc[lane][j] - mx) : 0.f;
            sc[lane][j] = e; sum += e;
        }
        float inv = any ? 1.f / sum : 0.f;
        for (int j = 0; j < NCP_; ++j) sc[lane][j] *= inv;
    }
    __syncthreads();
    for (int nt = 0; nt < HD_ / 16; ++nt) {
        v8f acc = V8F_ZERO;
        for (int kc = 0; kc < NCP_ / 32; ++kc) {
            v16h af = frag_a_f32(&sc[0][kc * 32], NCP_);
            v16h bf = frag_bt_f16(cvtb + (size_t)(nt * 16) * NCP_ + kc * 32, NCP_);
            acc = wmma32(af, bf, acc);
        }
#pragma unroll
        for (int r = 0; r < 8; ++r)
            oc[((size_t)(b * H_ + h) * S_ + m0 + mb + r) * HD_ + nt * 16 + ncol] = acc[r];
    }
}

// --------------------------------------------------- selected attention
__global__ void attn_sel(const _Float16* __restrict__ q16, const _Float16* __restrict__ k16,
                         const _Float16* __restrict__ vt, const int* __restrict__ sel,
                         float* __restrict__ os) {
    __shared__ float sc[16][LSEL_];
    int lane = threadIdx.x;
    int n = blockIdx.x >> 1, qt = blockIdx.x & 1;
    int y = blockIdx.y, g = y & 3, kv = (y >> 2) & 3, b = y >> 4;
    int h = kv * GQ_ + g;
    int s0 = n * BS_ + qt * 16;
    int ncol = lane & 15, kb0 = (lane >> 4) * 16, mb = (lane >> 4) * 8;
    int selv[TB_];
    const int* sp = sel + (size_t)((b * HKV_ + kv) * NB_ + n) * TB_;
#pragma unroll
    for (int t = 0; t < TB_; ++t) selv[t] = sp[t];
    const _Float16* qtile = q16 + (size_t)(b * S_ + s0) * (H_ * HD_) + h * HD_;
    const _Float16* vtb = vt + (size_t)(b * HKV_ + kv) * HD_ * S_;
    for (int nt = 0; nt < LSEL_ / 16; ++nt) {
        int t = nt >> 1, off = (nt & 1) * 16;
        int key = selv[t] * BS_ + off + ncol;
        const _Float16* kp = k16 + (size_t)(b * S_ + key) * (HKV_ * HD_) + kv * HD_;
        v8f acc = V8F_ZERO;
        for (int k0 = 0; k0 < HD_; k0 += 32) {
            v16h af = frag_a_f16(qtile + k0, H_ * HD_);
            v16h bf;
#pragma unroll
            for (int i = 0; i < 16; ++i) bf[i] = kp[k0 + kb0 + i];
            acc = wmma32(af, bf, acc);
        }
#pragma unroll
        for (int r = 0; r < 8; ++r) sc[mb + r][nt * 16 + ncol] = acc[r];
    }
    __syncthreads();
    if (lane < 16) {
        int s = s0 + lane;
        float mx = -INFINITY;
        for (int j = 0; j < LSEL_; ++j) {
            int pos = selv[j >> 5] * BS_ + (j & 31);
            float v = (pos <= s) ? sc[lane][j] * SCALE_ : NEGV;
            sc[lane][j] = v;
            if (v > mx) mx = v;
        }
        float sum = 0.f;
        for (int j = 0; j < LSEL_; ++j) { float e = expf(sc[lane][j] - mx); sc[lane][j] = e; sum += e; }
        float inv = 1.f / sum;
        for (int j = 0; j < LSEL_; ++j) sc[lane][j] *= inv;
    }
    __syncthreads();
    for (int nt = 0; nt < HD_ / 16; ++nt) {
        v8f acc = V8F_ZERO;
        for (int kc = 0; kc < LSEL_ / 32; ++kc) {
            v16h af = frag_a_f32(&sc[0][kc * 32], LSEL_);
            // B[k][n_d] = V[key(kc*32+k)][d] = vtb[d][key]; keys contiguous within chunk
            v16h bf = frag_bt_f16(vtb + (size_t)(nt * 16) * S_ + selv[kc] * BS_, S_);
            acc = wmma32(af, bf, acc);
        }
#pragma unroll
        for (int r = 0; r < 8; ++r)
            os[((size_t)(b * H_ + h) * S_ + s0 + mb + r) * HD_ + nt * 16 + ncol] = acc[r];
    }
}

// ---------------------------------------------------- windowed attention
__global__ void attn_win(const _Float16* __restrict__ q16, const _Float16* __restrict__ k16,
                         const _Float16* __restrict__ vt, float* __restrict__ ow) {
    __shared__ float sc[16][WK_];
    int lane = threadIdx.x;
    int n = blockIdx.x >> 1, qt = blockIdx.x & 1;
    int bh = blockIdx.y, b = bh >> 4, h = bh & 15, kv = h >> 2;
    int s0 = n * BS_ + qt * 16;
    int w0 = n * BS_ - WIN_;
    int ncol = lane & 15, kb0 = (lane >> 4) * 16, mb = (lane >> 4) * 8;
    const _Float16* qtile = q16 + (size_t)(b * S_ + s0) * (H_ * HD_) + h * HD_;
    const _Float16* vtb = vt + (size_t)(b * HKV_ + kv) * HD_ * S_;
    for (int nt = 0; nt < WK_ / 16; ++nt) {
        int pos = w0 + nt * 16 + ncol;
        int cl = pos < 0 ? 0 : (pos > S_ - 1 ? S_ - 1 : pos);
        const _Float16* kp = k16 + (size_t)(b * S_ + cl) * (HKV_ * HD_) + kv * HD_;
        v8f acc = V8F_ZERO;
        for (int k0 = 0; k0 < HD_; k0 += 32) {
            v16h af = frag_a_f16(qtile + k0, H_ * HD_);
            v16h bf;
#pragma unroll
            for (int i = 0; i < 16; ++i) bf[i] = kp[k0 + kb0 + i];
            acc = wmma32(af, bf, acc);
        }
#pragma unroll
        for (int r = 0; r < 8; ++r) sc[mb + r][nt * 16 + ncol] = acc[r];
    }
    __syncthreads();
    if (lane < 16) {
        int s = s0 + lane;
        float mx = -INFINITY;
        for (int j = 0; j < WK_; ++j) {
            int pos = w0 + j;
            float v = (pos >= 0 && pos <= s) ? sc[lane][j] * SCALE_ : NEGV;
            sc[lane][j] = v;
            if (v > mx) mx = v;
        }
        float sum = 0.f;
        for (int j = 0; j < WK_; ++j) { float e = expf(sc[lane][j] - mx); sc[lane][j] = e; sum += e; }
        float inv = 1.f / sum;
        for (int j = 0; j < WK_; ++j) sc[lane][j] *= inv;
    }
    __syncthreads();
    for (int nt = 0; nt < HD_ / 16; ++nt) {
        v8f acc = V8F_ZERO;
        for (int kc = 0; kc < WK_ / 32; ++kc) {
            v16h af = frag_a_f32(&sc[0][kc * 32], WK_);
            v16h bf;
#pragma unroll
            for (int i = 0; i < 16; ++i) {
                int pos = w0 + kc * 32 + kb0 + i;
                int cl = pos < 0 ? 0 : (pos > S_ - 1 ? S_ - 1 : pos);
                bf[i] = vtb[(size_t)(nt * 16 + ncol) * S_ + cl];  // masked probs are 0
            }
            acc = wmma32(af, bf, acc);
        }
#pragma unroll
        for (int r = 0; r < 8; ++r)
            ow[((size_t)(b * H_ + h) * S_ + s0 + mb + r) * HD_ + nt * 16 + ncol] = acc[r];
    }
}

// ------------------------------------------------------------ gated combine
__global__ void combine(const float* __restrict__ g, const float* __restrict__ oc,
                        const float* __restrict__ os, const float* __restrict__ ow,
                        _Float16* __restrict__ att) {
    size_t idx = (size_t)blockIdx.x * 256 + threadIdx.x;
    if (idx >= (size_t)B_ * S_ * H_ * HD_) return;
    int d = idx & 63;
    int h = (idx >> 6) & 15;
    int s = (idx >> 10) & 2047;
    int b = (int)(idx >> 21);
    size_t so = ((size_t)(b * H_ + h) * S_ + s) * HD_ + d;
    const float* gp = g + (size_t)(b * S_ + s) * (H_ * 3) + h * 3;
    att[idx] = (_Float16)(gp[0] * oc[so] + gp[1] * os[so] + gp[2] * ow[so]);
}

// ============================================================ launcher
extern "C" void kernel_launch(void* const* d_in, const int* in_sizes, int n_in,
                              void* d_out, int out_size, void* d_ws, size_t ws_size,
                              hipStream_t stream) {
    (void)in_sizes; (void)n_in; (void)out_size; (void)ws_size;
    const float* x    = (const float*)d_in[0];
    const float* ln1w = (const float*)d_in[1];
    const float* ln1b = (const float*)d_in[2];
    const float* wq   = (const float*)d_in[3];
    const float* bq   = (const float*)d_in[4];
    const float* wk   = (const float*)d_in[5];
    const float* bk   = (const float*)d_in[6];
    const float* wv   = (const float*)d_in[7];
    const float* bv   = (const float*)d_in[8];
    const float* wg   = (const float*)d_in[9];
    const float* bg   = (const float*)d_in[10];
    const float* wo   = (const float*)d_in[11];
    const float* bo   = (const float*)d_in[12];
    const float* ln2w = (const float*)d_in[13];
    const float* ln2b = (const float*)d_in[14];
    const float* w1   = (const float*)d_in[15];
    const float* b1   = (const float*)d_in[16];
    const float* w2   = (const float*)d_in[17];
    const float* b2   = (const float*)d_in[18];

    char* ws = (char*)d_ws;
    size_t off = 0;
    auto alloc = [&](size_t bytes) -> void* {
        void* p = ws + off;
        off += (bytes + 255) & ~(size_t)255;
        return p;
    };
    // transposed f16 weights BT[N][K]
    _Float16* WQt = (_Float16*)alloc((size_t)1024 * D_ * 2);
    _Float16* WKt = (_Float16*)alloc((size_t)256 * D_ * 2);
    _Float16* WVt = (_Float16*)alloc((size_t)256 * D_ * 2);
    _Float16* WGt = (_Float16*)alloc((size_t)48 * D_ * 2);
    _Float16* WOt = (_Float16*)alloc((size_t)D_ * 1024 * 2);
    _Float16* W1t = (_Float16*)alloc((size_t)4096 * D_ * 2);
    _Float16* W2t = (_Float16*)alloc((size_t)D_ * 4096 * 2);
    _Float16* H16 = (_Float16*)alloc((size_t)MROWS_ * D_ * 2);
    _Float16* Q16 = (_Float16*)alloc((size_t)MROWS_ * 1024 * 2);
    _Float16* K16 = (_Float16*)alloc((size_t)MROWS_ * 256 * 2);
    _Float16* V16 = (_Float16*)alloc((size_t)MROWS_ * 256 * 2);
    _Float16* VT  = (_Float16*)alloc((size_t)B_ * HKV_ * HD_ * S_ * 2);
    float*    G32 = (float*)alloc((size_t)MROWS_ * 48 * 4);
    _Float16* CK  = (_Float16*)alloc((size_t)B_ * HKV_ * NCP_ * HD_ * 2);
    _Float16* CVT = (_Float16*)alloc((size_t)B_ * HKV_ * HD_ * NCP_ * 2);
    float*    QSM = (float*)alloc((size_t)B_ * HKV_ * NB_ * HD_ * 4);
    float*    KBM = (float*)alloc((size_t)B_ * HKV_ * NB_ * HD_ * 4);
    int*      SEL = (int*)alloc((size_t)B_ * HKV_ * NB_ * TB_ * 4);
    float*    OC  = (float*)alloc((size_t)B_ * H_ * S_ * HD_ * 4);
    float*    OS  = (float*)alloc((size_t)B_ * H_ * S_ * HD_ * 4);
    float*    OW  = (float*)alloc((size_t)B_ * H_ * S_ * HD_ * 4);
    float*    X1  = (float*)alloc((size_t)MROWS_ * D_ * 4);
    // aliases (disjoint lifetimes): ATT reuses Q16, H2 reuses H16, MID reuses OC+OS
    _Float16* ATT = Q16;
    _Float16* H2  = H16;
    _Float16* MID = (_Float16*)OC;   // 4096*4096*2 == |OC|+|OS|

    // 1) weights -> transposed f16
    cvt_tr<<<dim3(32, 32), 256, 0, stream>>>(wq, WQt, D_, 1024);
    cvt_tr<<<dim3(32, 8), 256, 0, stream>>>(wk, WKt, D_, 256);
    cvt_tr<<<dim3(32, 8), 256, 0, stream>>>(wv, WVt, D_, 256);
    cvt_tr<<<dim3(32, 2), 256, 0, stream>>>(wg, WGt, D_, 48);
    cvt_tr<<<dim3(32, 32), 256, 0, stream>>>(wo, WOt, 1024, D_);
    cvt_tr<<<dim3(32, 128), 256, 0, stream>>>(w1, W1t, D_, 4096);
    cvt_tr<<<dim3(128, 32), 256, 0, stream>>>(w2, W2t, 4096, D_);

    // 2) LN1 and projections
    ln_f16<<<MROWS_, 256, 0, stream>>>(x, ln1w, ln1b, H16);
    gemm16<1><<<dim3(MROWS_ / 64, 16), 128, 0, stream>>>(H16, WQt, bq, MROWS_, 1024, D_, Q16, nullptr);
    gemm16<1><<<dim3(MROWS_ / 64, 4), 128, 0, stream>>>(H16, WKt, bk, MROWS_, 256, D_, K16, nullptr);
    gemm16<1><<<dim3(MROWS_ / 64, 4), 128, 0, stream>>>(H16, WVt, bv, MROWS_, 256, D_, V16, nullptr);
    gemm16<3><<<dim3(MROWS_ / 64, 1), 128, 0, stream>>>(H16, WGt, bg, MROWS_, 48, D_, G32, nullptr);
    v_transpose<<<(B_ * HKV_ * HD_ * S_) / 256, 256, 0, stream>>>(V16, VT);

    // 3) pooling / importance / top-k
    compress_kv<<<dim3(NCP_, B_ * HKV_), HD_, 0, stream>>>(K16, V16, CK, CVT);
    block_means<<<dim3(NB_, B_ * HKV_), HD_, 0, stream>>>(Q16, K16, QSM, KBM);
    imp_topk<<<B_ * HKV_, NB_, 0, stream>>>(QSM, KBM, SEL);

    // 4) attention branches
    attn_cmp<<<dim3(S_ / 16, B_ * H_), 32, 0, stream>>>(Q16, CK, CVT, OC);
    attn_sel<<<dim3(NB_ * 2, B_ * HKV_ * GQ_), 32, 0, stream>>>(Q16, K16, VT, SEL, OS);
    attn_win<<<dim3(NB_ * 2, B_ * H_), 32, 0, stream>>>(Q16, K16, VT, OW);

    // 5) gate combine + output projection (+residual x)
    combine<<<(B_ * S_ * H_ * HD_) / 256, 256, 0, stream>>>(G32, OC, OS, OW, ATT);
    gemm16<4><<<dim3(MROWS_ / 64, 16), 128, 0, stream>>>(ATT, WOt, bo, MROWS_, D_, 1024, X1, x);

    // 6) LN2 + MLP (+residual x1) -> d_out (f32)
    ln_f16<<<MROWS_, 256, 0, stream>>>(X1, ln2w, ln2b, H2);
    gemm16<2><<<dim3(MROWS_ / 64, 64), 128, 0, stream>>>(H2, W1t, b1, MROWS_, 4096, D_, MID, nullptr);
    gemm16<4><<<dim3(MROWS_ / 64, 16), 128, 0, stream>>>(MID, W2t, b2, MROWS_, D_, 4096, (float*)d_out, X1);
}
